// Structure_Decoder_3745211482437
// MI455X (gfx1250) — matrix operand order repaired and checked by reading, not compiled, not run on hardware
//
#include <hip/hip_runtime.h>
#include <hip/hip_bf16.h>
#include <stdint.h>

// ---------------------------------------------------------------------------
// Structure decoder: out = relu(GCNConv(x)) @ relu(GCNConv(x))^T
//   N = 16384 nodes, NHID = 64, E = 524288 edges, out = 16384 x 16384 f32.
// Dominant cost: 1 GiB output stream (~46 us at 23.3 TB/s). The 34 GFLOP
// H*H^T GEMM runs in f16 WMMA (v_wmma_f32_16x16x32_f16, K=64 -> 2 per tile);
// H (f16) is only 2 MiB -> lives in L2. Output uses non-temporal stores.
// ---------------------------------------------------------------------------

#define NN 16384      // nodes
#define NH 64         // hidden
#define NE 524288     // edges

typedef __attribute__((ext_vector_type(16))) _Float16 v16h;
typedef __attribute__((ext_vector_type(8)))  _Float16 v8h;
typedef __attribute__((ext_vector_type(8)))  float    v8f;

// ------------------------------ stage kernels ------------------------------

__global__ __launch_bounds__(256) void k_init_deg(float* __restrict__ deg) {
    int i = blockIdx.x * 256 + threadIdx.x;
    if (i < NN) deg[i] = 1.0f;  // self-loop contribution
}

__global__ __launch_bounds__(256) void k_deg(const long long* __restrict__ ei,
                                             float* __restrict__ deg) {
    int e = blockIdx.x * 256 + threadIdx.x;
    if (e < NE) {
        long long d = ei[NE + e];           // dst row of edge_index
        unsafeAtomicAdd(&deg[d], 1.0f);     // global_atomic_add_f32
    }
}

__global__ __launch_bounds__(256) void k_dinv(const float* __restrict__ deg,
                                              float* __restrict__ dinv) {
    int i = blockIdx.x * 256 + threadIdx.x;
    if (i < NN) {
        float d = deg[i];
        dinv[i] = (d > 0.0f) ? rsqrtf(d) : 0.0f;
    }
}

// h0 = x @ W  (16384x64 * 64x64). W cached in LDS; 4 rows per 256-thr block.
__global__ __launch_bounds__(256) void k_xw(const float* __restrict__ x,
                                            const float* __restrict__ W,
                                            float* __restrict__ h0) {
    __shared__ float Wl[NH * NH];           // 16 KB
    for (int i = threadIdx.x; i < NH * NH; i += 256) Wl[i] = W[i];
    __syncthreads();
    int row = blockIdx.x * 4 + (threadIdx.x >> 6);
    int col = threadIdx.x & 63;
    const float* xr = x + (size_t)row * NH;
    float s = 0.0f;
#pragma unroll
    for (int k = 0; k < NH; ++k) s = fmaf(xr[k], Wl[k * NH + col], s);
    h0[(size_t)row * NH + col] = s;
}

// acc[i] = h0[i] * dinv[i]^2   (self-loop message; also initializes acc)
__global__ __launch_bounds__(256) void k_self(const float* __restrict__ h0,
                                              const float* __restrict__ dinv,
                                              float* __restrict__ acc) {
    int idx = blockIdx.x * 256 + threadIdx.x;   // over NN*NH
    int node = idx >> 6;
    float di = dinv[node];
    acc[idx] = h0[idx] * di * di;
}

// edge scatter: acc[dst] += h0[src] * dinv[src]*dinv[dst]; 4 edges / block,
// 64 columns per edge so atomic addresses are contiguous per edge.
__global__ __launch_bounds__(256) void k_edge(const long long* __restrict__ ei,
                                              const float* __restrict__ h0,
                                              const float* __restrict__ dinv,
                                              float* __restrict__ acc) {
    int e   = blockIdx.x * 4 + (threadIdx.x >> 6);
    int col = threadIdx.x & 63;
    long long s = ei[e];
    long long d = ei[NE + e];
    float nrm = dinv[s] * dinv[d];
    float v = h0[(size_t)s * NH + col] * nrm;
    unsafeAtomicAdd(&acc[(size_t)d * NH + col], v);
}

// hf16 = (f16) relu(acc + b)
__global__ __launch_bounds__(256) void k_fin(const float* __restrict__ acc,
                                             const float* __restrict__ b,
                                             _Float16* __restrict__ hf) {
    int idx = blockIdx.x * 256 + threadIdx.x;   // over NN*NH
    int col = idx & 63;
    float v = acc[idx] + b[col];
    v = v > 0.0f ? v : 0.0f;
    hf[idx] = (_Float16)v;
}

// ------------------------------ H @ H^T (WMMA) -----------------------------
// A fragment (16x32 f16): lane m%16 holds row i0+m. Half-wave K split:
//   lanes 0-15  : K = kb+0..7   and kb+16..23
//   lanes 16-31 : K = kb+8..15  and kb+24..31
__device__ inline v16h load_afrag(const _Float16* __restrict__ row, int hi, int kb) {
    int o = kb + (hi ? 8 : 0);
    v8h lo8 = *(const v8h*)(row + o);
    v8h hi8 = *(const v8h*)(row + o + 16);
    return __builtin_shufflevector(lo8, hi8, 0, 1, 2, 3, 4, 5, 6, 7,
                                   8, 9, 10, 11, 12, 13, 14, 15);
}

// B fragment (32x16 f16, B[k][n] = H[j0+n][k]): lane n%16 holds column n,
// lanes 0-15 hold K=kb+0..7 and kb+16..23, lanes 16-31 hold K=kb+8..15 and kb+24..31.
__device__ inline v16h load_bfrag(const _Float16* __restrict__ row, int hi, int kb) {
    int o = kb + (hi ? 8 : 0);
    v8h lo8 = *(const v8h*)(row + o);
    v8h hi8 = *(const v8h*)(row + o + 16);
    return __builtin_shufflevector(lo8, hi8, 0, 1, 2, 3, 4, 5, 6, 7,
                                   8, 9, 10, 11, 12, 13, 14, 15);
}

// Block = 8 waves. Block computes a 16 x 512 slab: each wave keeps one A
// fragment pair (K=0..31, 32..63) in registers and sweeps 4 j-tiles.
__global__ __launch_bounds__(256) void k_gemm_hht(const _Float16* __restrict__ hf,
                                                  float* __restrict__ out) {
    const int lane = threadIdx.x & 31;
    const int wave = threadIdx.x >> 5;       // 0..7
    const int r    = lane & 15;
    const int hi   = (lane >> 4) & 1;

    const int i0 = blockIdx.y * 16;          // 1024 i-tiles
    const int jg = blockIdx.x * 512;         // 32 j-groups of 512 cols

    const _Float16* arow = hf + (size_t)(i0 + r) * NH;
    v16h a0 = load_afrag(arow, hi, 0);
    v16h a1 = load_afrag(arow, hi, 32);

#pragma unroll
    for (int t = 0; t < 4; ++t) {
        const int j0 = jg + wave * 64 + t * 16;
        const _Float16* brow = hf + (size_t)(j0 + r) * NH;
        v16h b0 = load_bfrag(brow, hi, 0);
        v16h b1 = load_bfrag(brow, hi, 32);

        v8f c = {};
        c = __builtin_amdgcn_wmma_f32_16x16x32_f16(false, a0, false, b0,
                                                   (short)0, c, false, false);
        c = __builtin_amdgcn_wmma_f32_16x16x32_f16(false, a1, false, b1,
                                                   (short)0, c, false, false);

        // C layout: VGPR q -> row (q or 8+q by half-wave), lane%16 -> col.
        const int rbase = i0 + (hi ? 8 : 0);
        float* ob = out + (size_t)rbase * NN + (size_t)(j0 + r);
#pragma unroll
        for (int q = 0; q < 8; ++q)
            __builtin_nontemporal_store(c[q], ob + (size_t)q * NN);
    }
}

// ------------------------------ launcher -----------------------------------

extern "C" void kernel_launch(void* const* d_in, const int* in_sizes, int n_in,
                              void* d_out, int out_size, void* d_ws, size_t ws_size,
                              hipStream_t stream) {
    const float*     x  = (const float*)d_in[0];
    const long long* ei = (const long long*)d_in[1];   // int64 edge_index [2, NE]
    const float*     W  = (const float*)d_in[2];
    const float*     b  = (const float*)d_in[3];
    float*           out = (float*)d_out;

    // workspace layout (~10.6 MB)
    float*     deg  = (float*)d_ws;              // NN
    float*     dinv = deg + NN;                  // NN
    float*     h0   = dinv + NN;                 // NN*NH
    float*     acc  = h0 + (size_t)NN * NH;      // NN*NH
    _Float16*  hf   = (_Float16*)(acc + (size_t)NN * NH);  // NN*NH halves

    k_init_deg<<<NN / 256, 256, 0, stream>>>(deg);
    k_deg<<<NE / 256, 256, 0, stream>>>(ei, deg);
    k_dinv<<<NN / 256, 256, 0, stream>>>(deg, dinv);
    k_xw<<<NN / 4, 256, 0, stream>>>(x, W, h0);
    k_self<<<(NN * NH) / 256, 256, 0, stream>>>(h0, dinv, acc);
    k_edge<<<NE / 4, 256, 0, stream>>>(ei, h0, dinv, acc);
    k_fin<<<(NN * NH) / 256, 256, 0, stream>>>(acc, b, hf);

    dim3 grid(NN / 512, NN / 16);   // (32, 1024)
    k_gemm_hht<<<grid, 256, 0, stream>>>(hf, out);
}
